// ESN_7438883357399
// MI455X (gfx1250) — compile-verified, matrix operand order checked by reading, not run on hardware
//
#include <hip/hip_runtime.h>
#include <hip/hip_bf16.h>
#include <cstdint>

// ---------------------------------------------------------------------------
// ESN forward for MI455X (gfx1250, wave32, WMMA + Tensor Data Mover).
// Sizes: VOCAB=50257, EMBED=256, RES=2048, SQV=225 (padded 240), B=4, T=512.
// GEMMs: v_wmma_f32_16x16x32_bf16, B-panels staged to LDS via tensor_load_to_lds.
// ---------------------------------------------------------------------------

typedef __bf16 bf16;
typedef __attribute__((ext_vector_type(16))) __bf16        v16bf;
typedef __attribute__((ext_vector_type(8)))  float         v8f;
typedef __attribute__((ext_vector_type(4)))  unsigned int  u32x4;
typedef __attribute__((ext_vector_type(8)))  int           i32x8;
typedef __attribute__((ext_vector_type(4)))  int           i32x4;

#define BATCH 4
#define TSEQ  512
#define NROW  (BATCH * TSEQ)   // 2048
#define EMB   256
#define RES   2048
#define SQV   225
#define SQVP  240              // padded to multiple of 16
#define VOCAB 50257

#define LDS_LD 136             // 128 + 4-DWORD pad per row -> bank spread

#if __has_builtin(__builtin_amdgcn_tensor_load_to_lds) && \
    __has_builtin(__builtin_amdgcn_s_wait_tensorcnt)
#define ESN_HAVE_TDM 1
#else
#define ESN_HAVE_TDM 0
#endif

// ---------------------------------------------------------------------------
// 16-bit WMMA operand fragment load (NT: both A[MxK] and B[NxK] row-major).
// ISA 16-bit 16x32 layout: lanes 0-15 -> row=lane, K={k0..k0+7, k0+16..k0+23};
// lanes 16-31 -> row=lane-16, K={k0+8..k0+15, k0+24..k0+31}.
// Two b128 loads per lane (global_load_b128 or ds_load_b128 by address space).
// ---------------------------------------------------------------------------
__device__ __forceinline__ v16bf ldfrag(const bf16* p, int ld) {
  int lane = threadIdx.x;                       // blockDim.x == 32 in GEMM kernels
  const bf16* q = p + (lane & 15) * ld + ((lane >> 4) << 3);
  v16bf f;
  __builtin_memcpy(&f, q, 16);
  __builtin_memcpy(reinterpret_cast<char*>(&f) + 16, q + 16, 16);
  return f;
}

#if ESN_HAVE_TDM
// ---------------------------------------------------------------------------
// TDM: DMA one 16(row) x 128(col) bf16 tile (row stride ld_elems) into LDS at
// byte address lds_addr, padding LDS by 4 DWORDs every 64 DWORDs (row -> 272B).
// D# layout per CDNA5 ISA ch.8: group0 = {flags, lds_addr, global_addr, type=2},
// group1 = {data_size=2B, pad cfg, tensor/tile dims, dim0 stride}.
// ---------------------------------------------------------------------------
__device__ __forceinline__ void tdm_load_tile16x128(const bf16* gsrc,
                                                    unsigned lds_addr,
                                                    int ld_elems) {
  unsigned long long ga = (unsigned long long)(uintptr_t)gsrc;
  u32x4 g0;
  g0[0] = 1u;                                            // count=1, user mode
  g0[1] = lds_addr;                                      // LDS byte address
  g0[2] = (unsigned)(ga & 0xffffffffu);                  // global addr [31:0]
  g0[3] = ((unsigned)(ga >> 32) & 0x01ffffffu)           // global addr [56:32]
          | (2u << 30);                                  // type = 2 ("image")
  i32x8 g1;
  g1[0] = (int)((1u << 16)        // data_size = 2 bytes
              | (1u << 20)        // pad_enable
              | (5u << 22)        // pad_interval: 64 DWORDs
              | (3u << 25));      // pad_amount: 4 DWORDs
  g1[1] = (int)((128u & 0xffffu) << 16);                 // tensor_dim0 lo16
  g1[2] = (int)((16u & 0xffffu) << 16);                  // dim0 hi | tensor_dim1 lo
  g1[3] = (int)(128u << 16);                             // dim1 hi | tile_dim0
  g1[4] = (int)16u;                                      // tile_dim1 | tile_dim2=0
  g1[5] = (int)(unsigned)ld_elems;                       // tensor_dim0_stride lo32
  g1[6] = 0;                                             // stride0 hi | stride1 lo
  g1[7] = 0;
  i32x4 gz = {0, 0, 0, 0};
#if defined(__clang_major__) && (__clang_major__ >= 23)
  i32x8 gz8 = {0, 0, 0, 0, 0, 0, 0, 0};
  __builtin_amdgcn_tensor_load_to_lds(g0, g1, gz, gz, gz8, 0);
#else
  __builtin_amdgcn_tensor_load_to_lds(g0, g1, gz, gz, 0);
#endif
}
#endif

// ---------------------------------------------------------------------------
// NT bf16 GEMM: C[M,N] = A[M,K] x B[N,K]^T (+bias). M mult of 128, N mult of
// 16, K mult of 128. Block (32,8): 8 waves stacked on M, one shared 16-col
// B-panel staged through double-buffered LDS by the Tensor Data Mover.
// ---------------------------------------------------------------------------
__global__ void gemm_nt_bf16_kernel(const bf16* __restrict__ A,
                                    const bf16* __restrict__ B,
                                    const float* __restrict__ bias,
                                    float* __restrict__ Cf,
                                    bf16* __restrict__ Cb,
                                    int K, int ldc) {
  __shared__ bf16 smem[2][16 * LDS_LD];
  const int wave = threadIdx.y;
  const int lane = threadIdx.x;
  const int m0 = (blockIdx.y * 8 + wave) * 16;
  const int n0 = blockIdx.x * 16;
  const int nk = K >> 7;                                 // K / 128

#if ESN_HAVE_TDM
  const unsigned lds0 = (unsigned)(uintptr_t)&smem[0][0];
  const unsigned ldsStride = 16u * LDS_LD * 2u;
  if (wave == 0) tdm_load_tile16x128(B + (size_t)n0 * K, lds0, K);
#else
  {
    const int tid = wave * 32 + lane;
    for (int i = tid; i < 16 * 128; i += 256) {
      int r = i >> 7, c = i & 127;
      smem[0][r * LDS_LD + c] = B[(size_t)(n0 + r) * K + c];
    }
  }
#endif

  v8f acc = {};
  for (int kc = 0; kc < nk; ++kc) {
    const int kbase = kc << 7;
#if ESN_HAVE_TDM
    if (wave == 0) {
      if (kc + 1 < nk) {          // kick next chunk's DMA, wait only for current
        tdm_load_tile16x128(B + (size_t)n0 * K + kbase + 128,
                            lds0 + (unsigned)((kc + 1) & 1) * ldsStride, K);
        __builtin_amdgcn_s_wait_tensorcnt(1);
      } else {
        __builtin_amdgcn_s_wait_tensorcnt(0);
      }
    }
#endif
    __syncthreads();

    // Preload all 4 A fragments of this 128-wide chunk (8 b128 loads in flight),
    // B fragments come from LDS (ds_load_b128).
    const bf16* ag = A + (size_t)m0 * K + kbase;
    v16bf a0 = ldfrag(ag +  0, K);
    v16bf a1 = ldfrag(ag + 32, K);
    v16bf a2 = ldfrag(ag + 64, K);
    v16bf a3 = ldfrag(ag + 96, K);
    const bf16* bs = &smem[kc & 1][0];
    v16bf b0 = ldfrag(bs +  0, LDS_LD);
    acc = __builtin_amdgcn_wmma_f32_16x16x32_bf16(false, a0, false, b0, (short)0, acc, false, false);
    v16bf b1 = ldfrag(bs + 32, LDS_LD);
    acc = __builtin_amdgcn_wmma_f32_16x16x32_bf16(false, a1, false, b1, (short)0, acc, false, false);
    v16bf b2 = ldfrag(bs + 64, LDS_LD);
    acc = __builtin_amdgcn_wmma_f32_16x16x32_bf16(false, a2, false, b2, (short)0, acc, false, false);
    v16bf b3 = ldfrag(bs + 96, LDS_LD);
    acc = __builtin_amdgcn_wmma_f32_16x16x32_bf16(false, a3, false, b3, (short)0, acc, false, false);

    __syncthreads();              // buffer may be overwritten next iteration

#if !ESN_HAVE_TDM
    if (kc + 1 < nk) {
      const int tid = wave * 32 + lane;
      for (int i = tid; i < 16 * 128; i += 256) {
        int r = i >> 7, c = i & 127;
        smem[(kc + 1) & 1][r * LDS_LD + c] =
            B[(size_t)(n0 + r) * K + kbase + 128 + c];
      }
    }
#endif
  }

  // C/D layout: VGPR r: lanes 0-15 -> (M=r, N=lane); lanes 16-31 -> (M=8+r, N=lane-16)
  const int col   = n0 + (lane & 15);
  const int rbase = (lane >> 4) << 3;
  const float bv  = bias ? bias[col] : 0.0f;
#pragma unroll
  for (int r = 0; r < 8; ++r) {
    const int m = m0 + rbase + r;
    float v = acc[r] + bv;
    if (Cf) Cf[(size_t)m * ldc + col] = v;
    if (Cb) Cb[(size_t)m * ldc + col] = (bf16)v;
  }
}

// ---------------------------------------------------------------------------
// One recurrence step: hnew = tanh(xin[:,t,:] + h x W_hh^T + b_hh).
// h padded to 16 rows (rows 4..15 stay zero). Block (32,4), grid 32 ->
// 128 waves, one 16(N)-column tile each, K=2048 -> 64 WMMAs per wave.
// Latency-bound; W_hh (8 MB bf16) stays L2-resident across the 512 launches.
// ---------------------------------------------------------------------------
__global__ void esn_step_kernel(const bf16* __restrict__ hin,   // [16 x RES] bf16
                                const bf16* __restrict__ Whh,   // [RES x RES] bf16
                                const float* __restrict__ xin,  // [NROW x RES] f32
                                const float* __restrict__ b_hh, // [RES]
                                float* __restrict__ states,     // [NROW x RES] f32
                                bf16*  __restrict__ states_b,   // [NROW x RES] bf16
                                bf16*  __restrict__ hout,       // [16 x RES] bf16
                                int t) {
  const int nt = blockIdx.x * blockDim.y + threadIdx.y;
  const int n0 = nt * 16;
  v8f acc = {};
#pragma unroll 4
  for (int k = 0; k < RES; k += 32) {
    v16bf a = ldfrag(hin + k, RES);
    v16bf b = ldfrag(Whh + (size_t)n0 * RES + k, RES);
    acc = __builtin_amdgcn_wmma_f32_16x16x32_bf16(false, a, false, b,
                                                  (short)0, acc, false, false);
  }
  const int lane = threadIdx.x;
  if (lane < 16) {                 // rows 0..7 live in lanes 0-15; only rows<4 valid
    const int col = n0 + lane;
    const float bb = b_hh[col];
#pragma unroll
    for (int r = 0; r < 4; ++r) {
      const size_t o = ((size_t)(r * TSEQ + t)) * RES + col;
      const float v = tanhf(acc[r] + xin[o] + bb);
      states[o]   = v;
      states_b[o] = (bf16)v;
      hout[(size_t)r * RES + col] = (bf16)v;
    }
  }
}

// ---------------------------------------------------------------------------
// Elementwise helpers
// ---------------------------------------------------------------------------
__global__ void cvt_bf16_kernel(const float* __restrict__ in, bf16* __restrict__ out, int n) {
  int i = blockIdx.x * blockDim.x + threadIdx.x;
  if (i < n) out[i] = (bf16)in[i];
}

// rows >= rows_in are zero-padded (grid.x = rows_out)
__global__ void cvt_pad_kernel(const float* __restrict__ in, bf16* __restrict__ out,
                               int rows_in, int cols) {
  int r = blockIdx.x;
  for (int c = threadIdx.x; c < cols; c += blockDim.x)
    out[(size_t)r * cols + c] = (r < rows_in) ? (bf16)in[(size_t)r * cols + c] : (bf16)0.0f;
}

__global__ void zero_bf16_kernel(bf16* __restrict__ p, int n) {
  int i = blockIdx.x * blockDim.x + threadIdx.x;
  if (i < n) p[i] = (bf16)0.0f;
}

// emb_b[n, e] = bf16(wte[idx[n], e]); grid 2048, block 256
__global__ void gather_emb_kernel(const int* __restrict__ idx, const float* __restrict__ wte,
                                  bf16* __restrict__ emb) {
  int n = blockIdx.x, e = threadIdx.x;
  emb[(size_t)n * EMB + e] = (bf16)wte[(size_t)idx[n] * EMB + e];
}

// ---------------------------------------------------------------------------
// log_softmax over 225 logits + per-row NLL. grid 2048, block 256.
// which==0 -> target index = t/225, which==1 -> t%225.
// ---------------------------------------------------------------------------
__global__ void loss_lp_kernel(const float* __restrict__ logits, const int* __restrict__ targets,
                               float* __restrict__ lp, float* __restrict__ nll,
                               int* __restrict__ cnt, int which) {
  __shared__ float red[256];
  const int n = blockIdx.x, j = threadIdx.x;
  const float x = (j < SQV) ? logits[(size_t)n * SQVP + j] : -3.0e38f;
  red[j] = x; __syncthreads();
  for (int s = 128; s > 0; s >>= 1) { if (j < s) red[j] = fmaxf(red[j], red[j + s]); __syncthreads(); }
  const float mx = red[0]; __syncthreads();
  red[j] = (j < SQV) ? __expf(x - mx) : 0.0f; __syncthreads();
  for (int s = 128; s > 0; s >>= 1) { if (j < s) red[j] += red[j + s]; __syncthreads(); }
  const float lse = mx + __logf(red[0]);
  if (j < SQV) lp[(size_t)n * SQVP + j] = x - lse;
  const int tgt  = targets[n];
  const bool msk = (tgt != -1);
  const int safe = msk ? tgt : 0;
  const int ti   = (which == 0) ? (safe / SQV) : (safe % SQV);
  if (j == ti)          nll[n] = msk ? (lse - x) : 0.0f;
  if (!msk && j == 0)   nll[n] = 0.0f;
  if (which == 0 && j == 0) cnt[n] = msk ? 1 : 0;
}

__global__ void loss_final_kernel(const float* __restrict__ nll1, const float* __restrict__ nll2,
                                  const int* __restrict__ cnt, float* __restrict__ out_loss) {
  __shared__ float s[256];
  __shared__ int   sc[256];
  const int j = threadIdx.x;
  float a = 0.0f; int c = 0;
  for (int i = j; i < NROW; i += 256) { a += nll1[i] + nll2[i]; c += cnt[i]; }
  s[j] = a; sc[j] = c; __syncthreads();
  for (int st = 128; st > 0; st >>= 1) { if (j < st) { s[j] += s[j + st]; sc[j] += sc[j + st]; } __syncthreads(); }
  if (j == 0) out_loss[0] = s[0] / (float)(sc[0] > 0 ? sc[0] : 1);
}

// h_n[0,b,r] = states[b, T-1, r]
__global__ void hn_kernel(const float* __restrict__ states, float* __restrict__ out) {
  int i = blockIdx.x * blockDim.x + threadIdx.x;
  if (i < BATCH * RES) {
    int b = i >> 11, r = i & (RES - 1);
    out[i] = states[((size_t)(b * TSEQ + TSEQ - 1)) * RES + r];
  }
}

// out[n, i*225+j] = lp1[n,i] + lp2[n,j], sliced to VOCAB. grid (2048, 7), block 256.
__global__ void combine_kernel(const float* __restrict__ lp1, const float* __restrict__ lp2,
                               float* __restrict__ out) {
  __shared__ float l2s[SQV];
  const int n = blockIdx.x, j = threadIdx.x;
  if (j < SQV) l2s[j] = lp2[(size_t)n * SQVP + j];
  __syncthreads();
  float* o = out + (size_t)n * VOCAB;
  const int i_end = min((int)(blockIdx.y * 32 + 32), 224);
  for (int i = blockIdx.y * 32; i < i_end; ++i) {
    const float a  = lp1[(size_t)n * SQVP + i];
    const int base = i * SQV;
    if (j < SQV && base + j < VOCAB) o[base + j] = a + l2s[j];
  }
}

// ---------------------------------------------------------------------------
// Host driver
// ---------------------------------------------------------------------------
extern "C" void kernel_launch(void* const* d_in, const int* in_sizes, int n_in,
                              void* d_out, int out_size, void* d_ws, size_t ws_size,
                              hipStream_t stream) {
  (void)in_sizes; (void)n_in; (void)out_size; (void)ws_size;
  const int*   idx     = (const int*)d_in[0];
  const int*   targets = (const int*)d_in[1];
  const float* wte     = (const float*)d_in[2];
  const float* W_in    = (const float*)d_in[3];
  const float* W_ih    = (const float*)d_in[4];
  const float* b_ih    = (const float*)d_in[5];
  const float* W_hh    = (const float*)d_in[6];
  const float* b_hh    = (const float*)d_in[7];
  const float* Wh1     = (const float*)d_in[8];
  const float* Wh2     = (const float*)d_in[9];
  float* out = (float*)d_out;

  char* ws = (char*)d_ws;
  size_t off = 0;
  auto alloc = [&](size_t bytes) -> void* {
    void* p = ws + off;
    off = (off + bytes + 255) & ~(size_t)255;
    return p;
  };
  bf16*  W_in_b   = (bf16*)alloc((size_t)RES * EMB * 2);
  bf16*  W_ih_b   = (bf16*)alloc((size_t)RES * RES * 2);
  bf16*  W_hh_b   = (bf16*)alloc((size_t)RES * RES * 2);
  bf16*  Wh1_b    = (bf16*)alloc((size_t)SQVP * RES * 2);
  bf16*  Wh2_b    = (bf16*)alloc((size_t)SQVP * RES * 2);
  bf16*  emb_b    = (bf16*)alloc((size_t)NROW * EMB * 2);
  bf16*  proj_b   = (bf16*)alloc((size_t)NROW * RES * 2);
  float* xin      = (float*)alloc((size_t)NROW * RES * 4);
  float* states   = (float*)alloc((size_t)NROW * RES * 4);
  bf16*  states_b = (bf16*)alloc((size_t)NROW * RES * 2);
  bf16*  hbuf     = (bf16*)alloc((size_t)2 * 16 * RES * 2);
  float* logits1  = (float*)alloc((size_t)NROW * SQVP * 4);
  float* logits2  = (float*)alloc((size_t)NROW * SQVP * 4);
  float* lp1      = (float*)alloc((size_t)NROW * SQVP * 4);
  float* lp2      = (float*)alloc((size_t)NROW * SQVP * 4);
  float* nll1     = (float*)alloc((size_t)NROW * 4);
  float* nll2     = (float*)alloc((size_t)NROW * 4);
  int*   cntb     = (int*)  alloc((size_t)NROW * 4);

  const size_t LOGITS = (size_t)NROW * VOCAB;

  // Weight conversions to bf16 (+ zero-pad row tail of Wh1/Wh2 to 240)
  cvt_bf16_kernel<<<(RES * EMB + 255) / 256, 256, 0, stream>>>(W_in, W_in_b, RES * EMB);
  cvt_bf16_kernel<<<(RES * RES + 255) / 256, 256, 0, stream>>>(W_ih, W_ih_b, RES * RES);
  cvt_bf16_kernel<<<(RES * RES + 255) / 256, 256, 0, stream>>>(W_hh, W_hh_b, RES * RES);
  cvt_pad_kernel<<<SQVP, 256, 0, stream>>>(Wh1, Wh1_b, SQV, RES);
  cvt_pad_kernel<<<SQVP, 256, 0, stream>>>(Wh2, Wh2_b, SQV, RES);
  zero_bf16_kernel<<<(2 * 16 * RES + 255) / 256, 256, 0, stream>>>(hbuf, 2 * 16 * RES);

  // Embedding gather -> bf16
  gather_emb_kernel<<<NROW, EMB, 0, stream>>>(idx, wte, emb_b);

  // proj = emb x W_in^T           [2048 x 256 x 2048] -> bf16
  gemm_nt_bf16_kernel<<<dim3(RES / 16, NROW / 128), dim3(32, 8), 0, stream>>>(
      emb_b, W_in_b, nullptr, nullptr, proj_b, EMB, RES);
  // xin = proj x W_ih^T + b_ih    [2048 x 2048 x 2048] -> f32
  gemm_nt_bf16_kernel<<<dim3(RES / 16, NROW / 128), dim3(32, 8), 0, stream>>>(
      proj_b, W_ih_b, b_ih, xin, nullptr, RES, RES);

  // Recurrence: 512 dependent steps, double-buffered bf16 h (16-row padded)
  for (int t = 0; t < TSEQ; ++t) {
    const bf16* hin = hbuf + (size_t)(t & 1) * 16 * RES;
    bf16* hout      = hbuf + (size_t)((t + 1) & 1) * 16 * RES;
    esn_step_kernel<<<dim3(32), dim3(32, 4), 0, stream>>>(
        hin, W_hh_b, xin, b_hh, states, states_b, hout, t);
  }

  // logits1/2 = states x Wh^T     [2048 x 2048 x 240] -> f32
  gemm_nt_bf16_kernel<<<dim3(SQVP / 16, NROW / 128), dim3(32, 8), 0, stream>>>(
      states_b, Wh1_b, nullptr, logits1, nullptr, RES, SQVP);
  gemm_nt_bf16_kernel<<<dim3(SQVP / 16, NROW / 128), dim3(32, 8), 0, stream>>>(
      states_b, Wh2_b, nullptr, logits2, nullptr, RES, SQVP);

  // log_softmax + per-row NLL, then deterministic reduction
  loss_lp_kernel<<<NROW, 256, 0, stream>>>(logits1, targets, lp1, nll1, cntb, 0);
  loss_lp_kernel<<<NROW, 256, 0, stream>>>(logits2, targets, lp2, nll2, cntb, 1);
  loss_final_kernel<<<1, 256, 0, stream>>>(nll1, nll2, cntb, out + LOGITS);
  hn_kernel<<<(BATCH * RES + 255) / 256, 256, 0, stream>>>(states, out + LOGITS + 1);

  // combined full-vocab log-prob logits (412 MB stream of stores)
  combine_kernel<<<dim3(NROW, 7), 256, 0, stream>>>(lp1, lp2, out);
}
